// CausalSelfAttention_78013785964722
// MI455X (gfx1250) — matrix operand order repaired and checked by reading, not compiled, not run on hardware
//
#include <hip/hip_runtime.h>
#include <hip/hip_bf16.h>
#include <math.h>

typedef __attribute__((ext_vector_type(16))) _Float16 v16h;
typedef __attribute__((ext_vector_type(8)))  float    v8f;
typedef unsigned int u32;
typedef unsigned long long u64;
typedef __attribute__((ext_vector_type(4))) u32 v4u;
typedef __attribute__((ext_vector_type(4))) int v4i;
typedef __attribute__((ext_vector_type(8))) int v8i;

struct alignas(16) H8 { _Float16 h[8]; };

#define BB 4
#define HH 16
#define TT 2048
#define CC 1024
#define DD 64
#define TP 2080            // TT + 32 padding rows for K / V-transposed
#define MM (BB*TT)         // 8192
#define NQKV (3*CC)        // 3072

#if defined(__AMDGCN__) && __has_builtin(__builtin_amdgcn_tensor_load_to_lds)
#define USE_TDM 1
#else
#define USE_TDM 0
#endif

// ---------- WMMA fragment loaders ----------
// A-operand (16x32 f16): lane<16 -> M=lane, K = {0..7, 16..23}; lane>=16 -> K = {8..15, 24..31}
__device__ inline v16h ld_a(const _Float16* row, int kbase, int half) {
    const H8* p0 = (const H8*)(row + kbase + half * 8);
    const H8* p1 = (const H8*)(row + kbase + half * 8 + 16);
    union { v16h v; H8 h[2]; } u;
    u.h[0] = *p0; u.h[1] = *p1;
    return u.v;
}
// B-operand (32x16 f16), row = N (K-contiguous storage): lane<16 -> K = {0..7, 16..23}; lane>=16 -> K = {8..15, 24..31}
__device__ inline v16h ld_b(const _Float16* row, int kbase, int half) {
    const H8* p0 = (const H8*)(row + kbase + half * 8);
    const H8* p1 = (const H8*)(row + kbase + half * 8 + 16);
    union { v16h v; H8 h[2]; } u;
    u.h[0] = *p0; u.h[1] = *p1;
    return u.v;
}
__device__ inline v8f wmma16(v16h a, v16h b, v8f c) {
    return __builtin_amdgcn_wmma_f32_16x16x32_f16(false, a, false, b, (short)0, c, false, false);
}
__device__ inline v8f zero8() {
    v8f z = {0.f,0.f,0.f,0.f,0.f,0.f,0.f,0.f};
    return z;
}

// ---------- Tensor Data Mover: 2D tile (rows x 32 f16) global -> LDS ----------
#if USE_TDM
__device__ inline void tdm_load_tile(u32 lds_off, const void* gptr, u32 tile_cols,
                                     u32 tile_rows, u64 row_stride_elems) {
    u64 ga = (u64)(__UINTPTR_TYPE__)gptr;
    v4u g0;
    g0.x = 1u;                                            // count=1 (valid user D#)
    g0.y = lds_off;                                       // LDS byte address of tile
    g0.z = (u32)ga;                                       // global_addr[31:0]
    g0.w = (u32)((ga >> 32) & 0x1FFFFFFu) | (2u << 30);   // global_addr[56:32] | type=2
    const u64 dim0 = 0x40000000ull;                       // huge tensor dims: no OOB clamping
    const u64 dim1 = 0x00100000ull;
    const u64 s0   = row_stride_elems;
    v8i g1;
    g1[0] = (int)(1u << 16);                              // data_size = 1 -> 2 bytes/elem
    g1[1] = (int)(((u32)(dim0 & 0xFFFFu)) << 16);         // tensor_dim0[15:0]
    g1[2] = (int)(((u32)((dim0 >> 16) & 0xFFFFu)) | (((u32)(dim1 & 0xFFFFu)) << 16));
    g1[3] = (int)(((u32)((dim1 >> 16) & 0xFFFFu)) | (tile_cols << 16)); // tile_dim0
    g1[4] = (int)(tile_rows & 0xFFFFu);                   // tile_dim1 (tile_dim2 = 0)
    g1[5] = (int)(u32)(s0 & 0xFFFFFFFFull);               // tensor_dim0_stride[31:0]
    g1[6] = (int)(u32)((s0 >> 32) & 0xFFFFull);           // tensor_dim0_stride[47:32]
    g1[7] = 0;
    v4i z4 = {0, 0, 0, 0};
#if __clang_major__ >= 23
    v8i z8 = {0, 0, 0, 0, 0, 0, 0, 0};
    __builtin_amdgcn_tensor_load_to_lds(g0, g1, z4, z4, z8, 0);
#else
    __builtin_amdgcn_tensor_load_to_lds(g0, g1, z4, z4, 0);
#endif
}
__device__ inline void tdm_wait() {
#if __has_builtin(__builtin_amdgcn_s_wait_tensorcnt)
    __builtin_amdgcn_s_wait_tensorcnt(0);
#else
    asm volatile("s_wait_tensorcnt 0" ::: "memory");
#endif
}
#endif

// ---------- prep kernels ----------
__global__ void cvt_f32_to_f16_kernel(const float* __restrict__ in, _Float16* __restrict__ out, int n) {
    int i = blockIdx.x * 256 + threadIdx.x;
    if (i < n) out[i] = (_Float16)in[i];
}
// Wt[n*K + k] = W[k*N + n]  (produce N-major, K-contiguous weights for B operand)
__global__ void transpose_cvt_kernel(const float* __restrict__ W, _Float16* __restrict__ Wt, int K, int N) {
    int i = blockIdx.x * 256 + threadIdx.x;
    if (i >= N * K) return;
    int n = i / K, k = i - n * K;
    Wt[i] = (_Float16)W[(size_t)k * N + n];
}

// ---------- WMMA GEMM: C[M,N] = A[M,K] @ Bt[N,K]^T, TDM-staged LDS double buffer ----------
template <bool F16OUT>
__global__ __launch_bounds__(256) void gemm_wmma_kernel(const _Float16* __restrict__ A,
                                                        const _Float16* __restrict__ Bt,
                                                        void* __restrict__ Cout,
                                                        int N, int K) {
    // smem[buf][0] = A tile 64x32, smem[buf][1] = B tile 64x32  (16 KB total)
    __shared__ __align__(16) _Float16 smem[2][2][64 * 32];

    const int tid  = threadIdx.x;
    const int wave = tid >> 5, lane = tid & 31;
    const int half = lane >> 4, lr = lane & 15;
    const int mb64 = blockIdx.y * 64;         // block row base
    const int nb64 = blockIdx.x * 64;         // block col base
    const int wm   = (wave >> 1) * 16;        // wave row offset in tile
    const int wn   = (wave & 1) * 32;         // wave col offset in tile

    auto stage = [&](int buf, int k0) {
#if USE_TDM
        if (tid < 32) {
            tdm_load_tile((u32)(__UINTPTR_TYPE__)&smem[buf][0][0],
                          A + (size_t)mb64 * K + k0, 32, 64, (u64)K);
            tdm_load_tile((u32)(__UINTPTR_TYPE__)&smem[buf][1][0],
                          Bt + (size_t)nb64 * K + k0, 32, 64, (u64)K);
        }
#else
        for (int i = tid; i < 64 * 32; i += 256) {
            int r = i >> 5, c = i & 31;
            smem[buf][0][i] = A [(size_t)(mb64 + r) * K + k0 + c];
            smem[buf][1][i] = Bt[(size_t)(nb64 + r) * K + k0 + c];
        }
#endif
    };
    auto publish = [&]() {
#if USE_TDM
        if (tid < 32) tdm_wait();
#endif
        __syncthreads();
    };

    v8f acc0 = zero8(), acc1 = zero8();
    stage(0, 0);
    for (int k0 = 0; k0 < K; k0 += 32) {
        int buf = (k0 >> 5) & 1;
        publish();                                  // buf ready for all waves
        if (k0 + 32 < K) stage(buf ^ 1, k0 + 32);   // overlap next tile with compute
        const _Float16* As = &smem[buf][0][0];
        const _Float16* Bs = &smem[buf][1][0];
        v16h a  = ld_a(As + (wm + lr) * 32, 0, half);
        v16h b0 = ld_b(Bs + (wn + lr) * 32, 0, half);
        v16h b1 = ld_b(Bs + (wn + 16 + lr) * 32, 0, half);
        acc0 = wmma16(a, b0, acc0);
        acc1 = wmma16(a, b1, acc1);
    }
#pragma unroll
    for (int i = 0; i < 8; ++i) {
        int row = mb64 + wm + 8 * half + i;
        size_t idx = (size_t)row * N + nb64 + wn + lr;
        if constexpr (F16OUT) {
            ((_Float16*)Cout)[idx]      = (_Float16)acc0[i];
            ((_Float16*)Cout)[idx + 16] = (_Float16)acc1[i];
        } else {
            ((float*)Cout)[idx]      = acc0[i];
            ((float*)Cout)[idx + 16] = acc1[i];
        }
    }
}

// ---------- RoPE + repack: QKV[b,t,3C] -> Qh[bh,t,d], Kh[bh,tP,d], Vt[bh,d,tP] ----------
__global__ void rope_pack_kernel(const _Float16* __restrict__ QKV,
                                 _Float16* __restrict__ Qh,
                                 _Float16* __restrict__ Kh,
                                 _Float16* __restrict__ Vt) {
    int idx = blockIdx.x * 256 + threadIdx.x;          // total = B*H*T*32
    int j  = idx & 31;
    int r  = idx >> 5;
    int t  = r & (TT - 1);
    int bh = r >> 11;
    int h  = bh & (HH - 1);
    int b  = bh >> 4;

    const _Float16* row = QKV + (size_t)(b * TT + t) * NQKV;
    int cq = h * DD + j;
    float q0 = (float)row[cq],            q1 = (float)row[cq + 32];
    float k0 = (float)row[CC + cq],       k1 = (float)row[CC + cq + 32];
    float v0 = (float)row[2 * CC + cq],   v1 = (float)row[2 * CC + cq + 32];

    float fr  = __expf(-9.210340371976184f * (float)j * (1.0f / 32.0f)); // 10000^(-j/32)
    float ang = (float)t * fr;
    float c = cosf(ang), s = sinf(ang);

    size_t qb = ((size_t)bh * TT + t) * DD + j;
    Qh[qb]      = (_Float16)(q0 * c - q1 * s);
    Qh[qb + 32] = (_Float16)(q0 * s + q1 * c);
    size_t kb = ((size_t)bh * TP + t) * DD + j;
    Kh[kb]      = (_Float16)(k0 * c - k1 * s);
    Kh[kb + 32] = (_Float16)(k0 * s + k1 * c);
    Vt[((size_t)bh * DD + j) * TP + t]      = (_Float16)v0;
    Vt[((size_t)bh * DD + j + 32) * TP + t] = (_Float16)v1;
}

// ---------- flash attention: one wave per 16-query tile ----------
__global__ __launch_bounds__(256) void flash_attn_kernel(const _Float16* __restrict__ Q,
                                                         const _Float16* __restrict__ K,
                                                         const _Float16* __restrict__ V,
                                                         _Float16* __restrict__ Y) {
    __shared__ __align__(16) _Float16 Pl[8][16 * 40];  // per-wave 16x32 P tile, stride 40 (16B aligned)

    const int tid  = threadIdx.x;
    const int wave = tid >> 5, lane = tid & 31;
    const int half = lane >> 4, lr = lane & 15;

    const int gw = blockIdx.x * 8 + wave;   // global wave = q-tile index
    const int bh = gw >> 7;                 // 128 tiles per (b,h)
    const int qt = gw & 127;
    const int b  = bh >> 4;
    const int h  = bh & 15;
    const int qbase = qt << 4;

    const _Float16* Qp = Q + ((size_t)bh * TT + qbase + lr) * DD;
    const _Float16* Kp = K + (size_t)bh * TP * DD;
    const _Float16* Vp = V + (size_t)bh * DD * TP;

    const v16h qf0 = ld_a(Qp, 0,  half);
    const v16h qf1 = ld_a(Qp, 32, half);

    float rmax[8], rsum[8];
    v8f Ov[4];
#pragma unroll
    for (int i = 0; i < 8; ++i) { rmax[i] = -3.0e38f; rsum[i] = 0.f; }
#pragma unroll
    for (int j = 0; j < 4; ++j) Ov[j] = zero8();

    _Float16* pl = &Pl[wave][0];

    for (int kb = 0; kb < qbase + 16; kb += 32) {
        // S = Q @ K^T for 16 queries x 32 keys (2 n-tiles, K-reduction over d=64 in two steps)
        const _Float16* k0row = Kp + (size_t)(kb + lr) * DD;
        const _Float16* k1row = Kp + (size_t)(kb + 16 + lr) * DD;
        v8f s0 = zero8(), s1 = zero8();
        s0 = wmma16(qf0, ld_b(k0row, 0,  half), s0);
        s0 = wmma16(qf1, ld_b(k0row, 32, half), s0);
        s1 = wmma16(qf0, ld_b(k1row, 0,  half), s1);
        s1 = wmma16(qf1, ld_b(k1row, 32, half), s1);

        // online softmax; D-layout: elem i holds row m = 8*half + i, col n = lr (+16 for s1)
#pragma unroll
        for (int i = 0; i < 8; ++i) {
            int mrow = qbase + 8 * half + i;
            float x0 = (kb + lr      <= mrow) ? s0[i] * 0.125f : -3.0e38f;
            float x1 = (kb + 16 + lr <= mrow) ? s1[i] * 0.125f : -3.0e38f;
            float mx = fmaxf(x0, x1);
#pragma unroll
            for (int m = 1; m < 16; m <<= 1) mx = fmaxf(mx, __shfl_xor(mx, m, 32));
            float nm   = fmaxf(rmax[i], mx);
            float corr = __expf(rmax[i] - nm);
            rmax[i] = nm;
            float p0 = __expf(x0 - nm);
            float p1 = __expf(x1 - nm);
            float ps = p0 + p1;
#pragma unroll
            for (int m = 1; m < 16; m <<= 1) ps += __shfl_xor(ps, m, 32);
            rsum[i] = rsum[i] * corr + ps;
            Ov[0][i] *= corr; Ov[1][i] *= corr; Ov[2][i] *= corr; Ov[3][i] *= corr;
            pl[(8 * half + i) * 40 + lr]      = (_Float16)p0;
            pl[(8 * half + i) * 40 + 16 + lr] = (_Float16)p1;
        }
        // own-wave LDS write -> read fence
        asm volatile("s_wait_dscnt 0" ::: "memory");

        // O += P @ V  (P as A operand 16x32, V^T rows as B operand)
        v16h pf = ld_a(pl + lr * 40, 0, half);
#pragma unroll
        for (int j = 0; j < 4; ++j) {
            const _Float16* vrow = Vp + (size_t)(j * 16 + lr) * TP + kb;
            Ov[j] = wmma16(pf, ld_b(vrow, 0, half), Ov[j]);
        }
    }

    // normalize + store Y[b, t, h*64 + d] as f16
#pragma unroll
    for (int j = 0; j < 4; ++j) {
#pragma unroll
        for (int i = 0; i < 8; ++i) {
            float o = Ov[j][i] / rsum[i];
            size_t yi = (size_t)(b * TT + qbase + 8 * half + i) * CC + h * DD + j * 16 + lr;
            Y[yi] = (_Float16)o;
        }
    }
}

extern "C" void kernel_launch(void* const* d_in, const int* in_sizes, int n_in,
                              void* d_out, int out_size, void* d_ws, size_t ws_size,
                              hipStream_t stream) {
    const float* x     = (const float*)d_in[0];   // [4,2048,1024]
    const float* w_qkv = (const float*)d_in[1];   // [1024,3072]
    const float* w_out = (const float*)d_in[2];   // [1024,1024]
    float* out = (float*)d_out;                   // [4,2048,1024] f32

    // workspace layout (f16 buffers), 256B aligned chunks
    char* ws = (char*)d_ws;
    auto take = [&](size_t bytes) { char* p = ws; ws += (bytes + 255) & ~(size_t)255; return p; };
    _Float16* Xh    = (_Float16*)take((size_t)MM * CC * 2);        // x in f16 (reused as Y later)
    _Float16* Wqkvt = (_Float16*)take((size_t)NQKV * CC * 2);      // w_qkv^T  [3072,1024]
    _Float16* Woutt = (_Float16*)take((size_t)CC * CC * 2);        // w_out^T  [1024,1024]
    _Float16* QKVh  = (_Float16*)take((size_t)MM * NQKV * 2);      // qkv      [8192,3072]
    _Float16* Qh    = (_Float16*)take((size_t)BB * HH * TT * DD * 2);
    _Float16* Kh    = (_Float16*)take((size_t)BB * HH * TP * DD * 2);
    _Float16* Vt    = (_Float16*)take((size_t)BB * HH * DD * TP * 2);
    _Float16* Y     = Xh;  // alias: Xh dead after QKV GEMM

    // 1) convert / transpose weights to f16 K-contiguous layouts
    {
        int n = MM * CC;
        cvt_f32_to_f16_kernel<<<(n + 255) / 256, 256, 0, stream>>>(x, Xh, n);
    }
    transpose_cvt_kernel<<<(NQKV * CC + 255) / 256, 256, 0, stream>>>(w_qkv, Wqkvt, CC, NQKV);
    transpose_cvt_kernel<<<(CC * CC + 255) / 256, 256, 0, stream>>>(w_out, Woutt, CC, CC);

    // 2) QKV = Xh @ Wqkv  (f16 out)
    gemm_wmma_kernel<true><<<dim3(NQKV / 64, MM / 64), 256, 0, stream>>>(Xh, Wqkvt, QKVh, NQKV, CC);

    // 3) RoPE + repack into Qh/Kh/Vt
    {
        int n = BB * HH * TT * 32;
        rope_pack_kernel<<<n / 256, 256, 0, stream>>>(QKVh, Qh, Kh, Vt);
    }

    // 4) causal flash attention -> Y (f16, [B,T,C])
    flash_attn_kernel<<<(BB * HH * (TT / 16)) / 8, 256, 0, stream>>>(Qh, Kh, Vt, Y);

    // 5) out = Y @ w_out  (f32 out)
    gemm_wmma_kernel<false><<<dim3(CC / 64, MM / 64), 256, 0, stream>>>(Y, Woutt, out, CC, CC);
}
